// VectorQuantizer_80298708566609
// MI455X (gfx1250) — matrix operand order repaired and checked by reading, not compiled, not run on hardware
//
#include <hip/hip_runtime.h>
#include <hip/hip_bf16.h>
#include <stdint.h>

// ---------------- problem constants ----------------
#define BATCH   32
#define CHN     256
#define HGT     64
#define WID     64
#define HW      (HGT * WID)          // 4096
#define NPTS    (BATCH * HW)         // 131072 spatial vectors
#define NE      1024                 // codebook entries
#define ED      256                  // embedding dim (== CHN)
#define NELEM   (BATCH * CHN * HW)   // 33554432 output elements
#define BETA    0.25f

// ---------------- tiling ----------------
#define TILE_S  128                  // spatial positions per workgroup
#define NWAVES  8                    // 256 threads
#define CHUNK   64                   // codes staged in LDS per iteration
#define NCHUNK  (NE / CHUNK)         // 16
#define KSTEP   32                   // K per wmma (bf16)
#define KITERS  (ED / KSTEP)         // 8
#define SZ      264                  // LDS row stride (bf16) for z tile; 264*2 = 528B, 16B aligned
#define SC      264                  // LDS row stride (bf16) for codebook chunk

// async double-buffered codebook staging (CDNA5 GLOBAL_LOAD_ASYNC_TO_LDS)
#if __has_builtin(__builtin_amdgcn_global_load_async_to_lds_b128)
#define HAS_ASYNC_LDS 1
#define NCBUF 2
#else
#define HAS_ASYNC_LDS 0
#define NCBUF 1
#endif

#define SMEM_BYTES ((TILE_S*SZ + NCBUF*CHUNK*SC) * 2 + NE * 4 + TILE_S * 4 + NWAVES * 4)

typedef __attribute__((ext_vector_type(16))) __bf16 v16bf;
typedef __attribute__((ext_vector_type(8)))  float  v8f;

#if HAS_ASYNC_LDS
typedef __attribute__((ext_vector_type(4))) int v4i;
typedef __attribute__((address_space(1))) v4i gv4i;   // global v4i32
typedef __attribute__((address_space(3))) v4i lv4i;   // LDS v4i32
#endif

union Frag16 {                       // 16 bf16 = 32 bytes = 8 dwords
    uint32_t u[8];
    v16bf    v;
};

__device__ __forceinline__ uint16_t f32_to_bf16_rne(float f) {
    uint32_t u = __float_as_uint(f);
    uint32_t r = u + 0x7FFFu + ((u >> 16) & 1u);
    return (uint16_t)(r >> 16);
}

__device__ __forceinline__ void wait_async0() {
#if HAS_ASYNC_LDS
#if __has_builtin(__builtin_amdgcn_s_wait_asynccnt)
    __builtin_amdgcn_s_wait_asynccnt(0);
#else
    asm volatile("s_wait_asynccnt 0x0" ::: "memory");
#endif
#endif
}

// A-fragment loader: 16-bit A 16x32 layout (ISA 7.12.2)
//   lane n = row M, lane-half kh: VGPR0-3 hold K = kh*8+0..7,
//   VGPR4-7 hold K = kh*8+16..23  -> two 16B contiguous LDS reads.
__device__ __forceinline__ void ldfrag(Frag16& f, const uint16_t* row, int k) {
    const uint16_t* p = row + k * KSTEP;
    uint4 lo = *(const uint4*)p;          // K kh*8 + 0..7
    uint4 hi = *(const uint4*)(p + 16);   // K kh*8 + 16..23
    f.u[0] = lo.x; f.u[1] = lo.y; f.u[2] = lo.z; f.u[3] = lo.w;
    f.u[4] = hi.x; f.u[5] = hi.y; f.u[6] = hi.z; f.u[7] = hi.w;
}

// ---------------------------------------------------------------------------
// Pass A: codebook -> bf16 copy in workspace + per-code squared norms
// ---------------------------------------------------------------------------
__global__ void vq_prep_kernel(const float* __restrict__ cb,
                               uint16_t*    __restrict__ cb_bf,
                               float*       __restrict__ norms) {
    const int j = blockIdx.x;
    const int t = threadIdx.x;
    float v = cb[j * ED + t];
    cb_bf[j * ED + t] = f32_to_bf16_rne(v);
    float s = v * v;
#pragma unroll
    for (int o = 16; o > 0; o >>= 1) s += __shfl_xor(s, o, 32);
    __shared__ float sred[NWAVES];
    if ((t & 31) == 0) sred[t >> 5] = s;
    __syncthreads();
    if (t == 0) {
        float tot = 0.f;
#pragma unroll
        for (int i = 0; i < NWAVES; ++i) tot += sred[i];
        norms[j] = tot;
    }
}

__global__ void vq_zero_loss_kernel(float* loss) { *loss = 0.f; }

// ---------------------------------------------------------------------------
// Main kernel: distances via v_wmma_f32_16x16x32_bf16, argmin, gather, loss.
//   D = A x B + C,  A = code tile (16x32 bf16), B = z tile (32x16 bf16),
//   C initialized to -0.5*||e||^2  ->  acc = -0.5||e||^2 + <e,z>  (argmax).
// ---------------------------------------------------------------------------
__global__ void vq_main_kernel(const float*    __restrict__ z,
                               const float*    __restrict__ cb_f32,
                               const uint16_t* __restrict__ cb_bf,
                               const float*    __restrict__ norms,
                               float*          __restrict__ out,
                               float*          __restrict__ loss) {
    extern __shared__ __align__(16) char smem[];
    uint16_t* z_lds    = (uint16_t*)smem;                       // TILE_S x SZ
    uint16_t* cb_lds   = z_lds + TILE_S * SZ;                   // NCBUF x CHUNK x SC
    float*    norm_lds = (float*)(cb_lds + NCBUF * CHUNK * SC); // NE
    int*      best_lds = (int*)(norm_lds + NE);                 // TILE_S
    float*    red_lds  = (float*)(best_lds + TILE_S);           // NWAVES

    const int t    = threadIdx.x;
    const int wave = t >> 5;
    const int lane = t & 31;
    const int n    = lane & 15;      // column (spatial) within 16x16 tile
    const int kh   = lane >> 4;      // lane-half selects K sub-range

    const int s0  = blockIdx.x * TILE_S;   // global spatial base
    const int bi  = s0 >> 12;              // image index  (s0 / 4096)
    const int hw0 = s0 & (HW - 1);         // offset inside image plane

    const size_t gimg = (size_t)bi * CHN * HW;

#if HAS_ASYNC_LDS
    // ---- prologue: kick off async DMA of chunk 0 into buffer 0 ----
    {
        const uint16_t* src = cb_bf;               // chunk 0
        uint16_t*       dst = cb_lds;              // buffer 0
#pragma unroll
        for (int r = 0; r < 8; ++r) {
            int idx  = r * 256 + t;                // [0, 2048) x 16B = 32 KB
            int code = idx >> 4;                   // [0, 128)
            int off  = (idx & 15) << 3;            // bf16 offset in row
            __builtin_amdgcn_global_load_async_to_lds_b128(
                (gv4i*)(src + code * ED + off),
                (lv4i*)(dst + code * SC + off), 0, 0);
        }
    }
#endif

    // ---- stage z tile:  [B,C,H,W] global  ->  LDS [s][c] bf16 (transpose) ----
#pragma unroll 4
    for (int it = 0; it < 64; ++it) {
        int idx = it * 256 + t;            // [0, 16384)
        int s   = idx & (TILE_S - 1);
        int cp  = idx >> 7;                // [0, 128)
        int c0  = cp << 1;
        float v0 = z[gimg + (size_t)c0 * HW + hw0 + s];
        float v1 = z[gimg + (size_t)(c0 + 1) * HW + hw0 + s];
        uint32_t pk = (uint32_t)f32_to_bf16_rne(v0) |
                      ((uint32_t)f32_to_bf16_rne(v1) << 16);
        *(uint32_t*)(z_lds + s * SZ + c0) = pk;
    }
    // ---- stage all code norms once (4 KB) ----
#pragma unroll
    for (int i = 0; i < NE / 256; ++i) norm_lds[i * 256 + t] = norms[i * 256 + t];

    __syncthreads();

    // ---- load this wave's B-panel (z fragments) for all 8 K-steps ----
    // B layout (32x16 bf16): lane n = column, lane-half kh picks K range,
    // VGPR0-3 hold K = kh*8+0..7, VGPR4-7 hold K = kh*8+16..23 (matches A).
    Frag16 bfr[KITERS];
    {
        const uint16_t* zrow = z_lds + (wave * 16 + n) * SZ;
#pragma unroll
        for (int k = 0; k < KITERS; ++k) {
            const uint16_t* p = zrow + k * KSTEP + kh * 8;
            uint4 a = *(const uint4*)p;
            uint4 b = *(const uint4*)(p + 16);
            bfr[k].u[0] = a.x; bfr[k].u[1] = a.y; bfr[k].u[2] = a.z; bfr[k].u[3] = a.w;
            bfr[k].u[4] = b.x; bfr[k].u[5] = b.y; bfr[k].u[6] = b.z; bfr[k].u[7] = b.w;
        }
    }

    float bestv = -3.402823466e+38f;
    int   besti = 0;

    // ---- loop over codebook in chunks of CHUNK codes ----
    for (int chunk = 0; chunk < NCHUNK; ++chunk) {
#if HAS_ASYNC_LDS
        wait_async0();       // my async copies into buf[cur] are in LDS
        __syncthreads();     // everyone's copies visible; prev reads of other buf done
        const int cur = chunk & 1;
        if (chunk + 1 < NCHUNK) {   // DMA next chunk into the other buffer
            const uint16_t* src = cb_bf + (size_t)(chunk + 1) * CHUNK * ED;
            uint16_t*       dst = cb_lds + (cur ^ 1) * (CHUNK * SC);
#pragma unroll
            for (int r = 0; r < 8; ++r) {
                int idx  = r * 256 + t;
                int code = idx >> 4;
                int off  = (idx & 15) << 3;
                __builtin_amdgcn_global_load_async_to_lds_b128(
                    (gv4i*)(src + code * ED + off),
                    (lv4i*)(dst + code * SC + off), 0, 0);
            }
        }
        const uint16_t* cbuf = cb_lds + cur * (CHUNK * SC);
#else
        __syncthreads();     // previous chunk's wmma reads done before restage
        {                    // sync staging fallback
            const uint16_t* src = cb_bf + (size_t)chunk * CHUNK * ED;
            if (chunk + 1 < NCHUNK)
                __builtin_prefetch(cb_bf + (size_t)(chunk + 1) * CHUNK * ED, 0, 1);
#pragma unroll 4
            for (int it = 0; it < 32; ++it) {
                int idx  = it * 256 + t;       // [0, 8192) dwords
                int code = idx >> 7;           // [0, 64)
                int c0   = (idx & 127) << 1;
                uint32_t pk = *(const uint32_t*)(src + code * ED + c0);
                *(uint32_t*)(cb_lds + code * SC + c0) = pk;
            }
        }
        __syncthreads();
        const uint16_t* cbuf = cb_lds;
#endif

        // ---- 4 column tiles processed as 2 pairs: two independent wmma
        //      chains + A-fragment loads software-pipelined one K-step ahead.
#pragma unroll
        for (int cth = 0; cth < 2; ++cth) {
            const int ctA  = cth * 2;
            const int ctB  = ctA + 1;
            const int cbA  = chunk * CHUNK + ctA * 16 + kh * 8;  // codes (chain A)
            const int cbB  = chunk * CHUNK + ctB * 16 + kh * 8;  // codes (chain B)

            v8f acc0, acc1;
#pragma unroll
            for (int r = 0; r < 8; ++r) acc0[r] = -0.5f * norm_lds[cbA + r];
#pragma unroll
            for (int r = 0; r < 8; ++r) acc1[r] = -0.5f * norm_lds[cbB + r];

            const uint16_t* rowA = cbuf + (ctA * 16 + n) * SC + kh * 8;
            const uint16_t* rowB = cbuf + (ctB * 16 + n) * SC + kh * 8;

            Frag16 a0, a1, p0, p1;
            ldfrag(a0, rowA, 0);
            ldfrag(a1, rowB, 0);
#pragma unroll
            for (int k = 0; k < KITERS; ++k) {
                if (k + 1 < KITERS) {          // prefetch next K-step fragments
                    ldfrag(p0, rowA, k + 1);
                    ldfrag(p1, rowB, k + 1);
                }
                acc0 = __builtin_amdgcn_wmma_f32_16x16x32_bf16(
                        false, a0.v, false, bfr[k].v, (short)0, acc0, false, false);
                acc1 = __builtin_amdgcn_wmma_f32_16x16x32_bf16(
                        false, a1.v, false, bfr[k].v, (short)0, acc1, false, false);
                a0 = p0;
                a1 = p1;
            }
            // argmax update (== argmin of distance); '>' keeps earliest index
#pragma unroll
            for (int r = 0; r < 8; ++r) {
                float v = acc0[r];
                if (v > bestv) { bestv = v; besti = cbA + r; }
            }
#pragma unroll
            for (int r = 0; r < 8; ++r) {
                float v = acc1[r];
                if (v > bestv) { bestv = v; besti = cbB + r; }
            }
        }
    }

    // combine lane halves (lane l holds codes with (code%16)<8, l+16 the rest)
    {
        float ov = __shfl_xor(bestv, 16, 32);
        int   oi = __shfl_xor(besti, 16, 32);
        if (ov > bestv || (ov == bestv && oi < besti)) { bestv = ov; besti = oi; }
        if (lane < 16) best_lds[wave * 16 + lane] = besti;
    }
    __syncthreads();

    // ---- output: gather f32 code rows, write [B,C,H,W], accumulate loss ----
    float lsum = 0.f;
#pragma unroll 4
    for (int it = 0; it < 128; ++it) {
        int e = it * 256 + t;            // [0, 32768)
        int s = e & (TILE_S - 1);
        int c = e >> 7;                  // [0, 256)
        int code = best_lds[s];
        float q  = cb_f32[(size_t)code * ED + c];
        size_t g = gimg + (size_t)c * HW + hw0 + s;
        float d  = q - z[g];
        lsum += d * d;
        out[g] = q;
    }
#pragma unroll
    for (int o = 16; o > 0; o >>= 1) lsum += __shfl_xor(lsum, o, 32);
    if (lane == 0) red_lds[wave] = lsum;
    __syncthreads();
    if (t == 0) {
        float tot = 0.f;
#pragma unroll
        for (int i = 0; i < NWAVES; ++i) tot += red_lds[i];
        // codebook_loss = (1 + BETA) * mean((z_q - z)^2)
        atomicAdd(loss, tot * ((1.0f + BETA) / (float)NELEM));
    }
}

// ---------------------------------------------------------------------------
// launch
// ---------------------------------------------------------------------------
extern "C" void kernel_launch(void* const* d_in, const int* in_sizes, int n_in,
                              void* d_out, int out_size, void* d_ws, size_t ws_size,
                              hipStream_t stream) {
    const float* z      = (const float*)d_in[0];  // [32,256,64,64]
    const float* cb_f32 = (const float*)d_in[1];  // [1024,256]

    float* out  = (float*)d_out;                  // 33554432 + 1
    float* loss = out + (size_t)NELEM;

    uint16_t* cb_bf = (uint16_t*)d_ws;                                 // 512 KB bf16 codebook
    float*    norms = (float*)((char*)d_ws + (size_t)NE * ED * 2);     // 4 KB norms

    vq_prep_kernel<<<NE, 256, 0, stream>>>(cb_f32, cb_bf, norms);
    vq_zero_loss_kernel<<<1, 1, 0, stream>>>(loss);
    vq_main_kernel<<<NPTS / TILE_S, NWAVES * 32, SMEM_BYTES, stream>>>(
        z, cb_f32, cb_bf, norms, out, loss);
}
